// UNSATMinimizer_29643864277125
// MI455X (gfx1250) — matrix operand order repaired and checked by reading, not compiled, not run on hardware
//
#include <hip/hip_runtime.h>
#include <hip/hip_fp16.h>

typedef __attribute__((ext_vector_type(16))) _Float16 v16h;
typedef __attribute__((ext_vector_type(8)))  _Float16 v8h;
typedef __attribute__((ext_vector_type(8)))  float    v8f;

#define NV 24576
#define NC 98304
#define NE (3*NC)
#define RNDS 4
#define EPSN 1e-6f

__device__ __forceinline__ float sigf(float x){ return 1.0f/(1.0f+__expf(-x)); }
__device__ __forceinline__ float sppf(float x){ return fmaxf(x,0.0f)+log1pf(__expf(-fabsf(x))); }

// ---------------------------------------------------------------- utility
__global__ void k_fill(float* __restrict__ p, float v, size_t n){
  for(size_t i=(size_t)blockIdx.x*blockDim.x+threadIdx.x; i<n; i+=(size_t)gridDim.x*blockDim.x) p[i]=v;
}

// f32 -> f16 weight convert with optional zero row padding (Rpad >= R)
__global__ void k_w2h(const float* __restrict__ s, _Float16* __restrict__ d, int R,int Rpad,int C){
  size_t n=(size_t)Rpad*C;
  for(size_t i=(size_t)blockIdx.x*blockDim.x+threadIdx.x; i<n; i+=(size_t)gridDim.x*blockDim.x){
    int r=(int)(i/C);
    d[i] = (r<R) ? (_Float16)s[i] : (_Float16)0.0f;
  }
}

// ---------------------------------------------------------------- degrees
__global__ void k_deg(const int* __restrict__ el, float* __restrict__ deg){
  for(size_t e=(size_t)blockIdx.x*blockDim.x+threadIdx.x; e<NE; e+=(size_t)gridDim.x*blockDim.x)
    atomicAdd(&deg[el[e]], 1.0f);
}
__global__ void k_degw(const float* __restrict__ deg, float* __restrict__ degw, float* __restrict__ vdw){
  for(size_t l=(size_t)blockIdx.x*blockDim.x+threadIdx.x; l<(size_t)2*NV; l+=(size_t)gridDim.x*blockDim.x){
    degw[l]=rsqrtf(fmaxf(deg[l],1.0f));
    if(l<NV) vdw[l]=4.0f*rsqrtf(fmaxf(deg[l]+deg[l+NV],1.0f));
  }
}

// ---------------------------------------------------------------- input builders
// X[M,160] = [state(128) | noise(4) | zeros(28)]  (f16)
__global__ void k_build_xin(const float* __restrict__ st, const float* __restrict__ nz,
                            _Float16* __restrict__ X, int M){
  size_t n=(size_t)M*160;
  for(size_t i=(size_t)blockIdx.x*blockDim.x+threadIdx.x; i<n; i+=(size_t)gridDim.x*blockDim.x){
    int r=(int)(i/160), j=(int)(i%160);
    float v = (j<128) ? st[(size_t)r*128+j] : (j<132 ? nz[(size_t)r*4+(j-128)] : 0.0f);
    X[i]=(_Float16)v;
  }
}

// csum[clause] += softplus(+-vq[var])
__global__ void k_scatter_lits(const int* __restrict__ el, const int* __restrict__ ec,
                               const float* __restrict__ vq, float* __restrict__ csum){
  size_t n=(size_t)NE*128;
  for(size_t i=(size_t)blockIdx.x*blockDim.x+threadIdx.x; i<n; i+=(size_t)gridDim.x*blockDim.x){
    int e=(int)(i>>7), j=(int)(i&127);
    int lit=el[e], c=ec[e];
    float q = (lit<NV) ? vq[(size_t)lit*128+j] : -vq[(size_t)(lit-NV)*128+j];
    atomicAdd(&csum[(size_t)c*128+j], sppf(q));
  }
}

// reads csum (in-place -> dcsum), cq, clauses ; writes X_cm[NC,384] f16 and dcsum
__global__ void k_clause_mid(float* __restrict__ csum_dcsum, const float* __restrict__ cq,
                             const float* __restrict__ cls, _Float16* __restrict__ Xcm){
  const float invMN = 1.0f/((float)NC*128.0f);
  size_t n=(size_t)NC*128;
  for(size_t i=(size_t)blockIdx.x*blockDim.x+threadIdx.x; i<n; i+=(size_t)gridDim.x*blockDim.x){
    int c=(int)(i>>7), j=(int)(i&127);
    float cs = csum_dcsum[i];
    float s  = sigf(cq[i]);
    float el = __expf(-cs);
    float loss = el*s;
    Xcm[(size_t)c*384      +j]=(_Float16)cls[i];
    Xcm[(size_t)c*384+128  +j]=(_Float16)(4.0f*loss);
    Xcm[(size_t)c*384+256  +j]=(_Float16)(loss*(1.0f-s)*invMN);
    csum_dcsum[i] = -loss*invMN;
  }
}

// dst[didx[e],j] += src[sidx[e]*sld + j]
__global__ void k_scatter_gather(const int* __restrict__ sidx, const int* __restrict__ didx,
                                 const float* __restrict__ src, int sld, float* __restrict__ dst){
  size_t n=(size_t)NE*128;
  for(size_t i=(size_t)blockIdx.x*blockDim.x+threadIdx.x; i<n; i+=(size_t)gridDim.x*blockDim.x){
    int e=(int)(i>>7), j=(int)(i&127);
    atomicAdd(&dst[(size_t)didx[e]*128+j], src[(size_t)sidx[e]*sld+j]);
  }
}

// X_ug[NV,512] = [vgrad | variables | vloss_pos*dw | vloss_neg*dw]  (f16)
__global__ void k_build_ug(const float* __restrict__ dlits, const float* __restrict__ vq,
                           const float* __restrict__ vdw, const float* __restrict__ vars,
                           const float* __restrict__ vloss, const float* __restrict__ degw,
                           _Float16* __restrict__ Xug){
  size_t n=(size_t)NV*128;
  for(size_t i=(size_t)blockIdx.x*blockDim.x+threadIdx.x; i<n; i+=(size_t)gridDim.x*blockDim.x){
    int v=(int)(i>>7), j=(int)(i&127);
    float s = sigf(vq[i]);
    float g = (dlits[i]*s - dlits[(size_t)NV*128+i]*(1.0f-s)) * vdw[v];
    Xug[(size_t)v*512      +j]=(_Float16)g;
    Xug[(size_t)v*512+128  +j]=(_Float16)vars[i];
    Xug[(size_t)v*512+256  +j]=(_Float16)(vloss[i]*degw[v]);
    Xug[(size_t)v*512+384  +j]=(_Float16)(vloss[(size_t)NV*128+i]*degw[NV+v]);
  }
}

// ---------------------------------------------------------------- pair_norm
__global__ void k_pn_reduce(const float* __restrict__ X, int M, int ld, int col0,
                            float* __restrict__ colsum, float* __restrict__ sumsq){
  __shared__ float sh[128];
  int j=threadIdx.x;               // 128 threads
  float s=0.f, ss=0.f;
  for(int r=blockIdx.x; r<M; r+=gridDim.x){
    float x = X[(size_t)r*ld + col0 + j];
    s+=x; ss+=x*x;
  }
  atomicAdd(&colsum[j], s);
  sh[j]=ss; __syncthreads();
  for(int o=64;o>0;o>>=1){ if(j<o) sh[j]+=sh[j+o]; __syncthreads(); }
  if(j==0) atomicAdd(sumsq, sh[0]);
}
__global__ void k_pn_final(const float* __restrict__ colsum, const float* __restrict__ sumsq,
                           float invM, float* __restrict__ ms){
  __shared__ float sh[128];
  int j=threadIdx.x;
  float mu = colsum[j]*invM;
  ms[j]=mu;
  sh[j]=mu*mu; __syncthreads();
  for(int o=64;o>0;o>>=1){ if(j<o) sh[j]+=sh[j+o]; __syncthreads(); }
  if(j==0){
    float msq = (*sumsq)*invM*(1.0f/128.0f) - sh[0]*(1.0f/128.0f);
    ms[128]=rsqrtf(msq + EPSN);
  }
}
// clauses = pn(CD[:,128:256])*0.25 + 0.1*clauses ; also emit f16 copy for co-MLP
__global__ void k_apply_clause(const float* __restrict__ CD, const float* __restrict__ ms,
                               float* __restrict__ cls, _Float16* __restrict__ Xco){
  size_t n=(size_t)NC*128;
  for(size_t i=(size_t)blockIdx.x*blockDim.x+threadIdx.x; i<n; i+=(size_t)gridDim.x*blockDim.x){
    int c=(int)(i>>7), j=(int)(i&127);
    float x = CD[(size_t)c*256 + 128 + j];
    float y = (x - ms[j])*ms[128]*0.25f + 0.1f*cls[i];
    cls[i]=y; Xco[i]=(_Float16)y;
  }
}
__global__ void k_apply_var(const float* __restrict__ V, const float* __restrict__ ms,
                            float* __restrict__ vars){
  size_t n=(size_t)NV*128;
  for(size_t i=(size_t)blockIdx.x*blockDim.x+threadIdx.x; i<n; i+=(size_t)gridDim.x*blockDim.x){
    int j=(int)(i&127);
    vars[i] = (V[i]-ms[j])*ms[128]*0.25f + 0.1f*vars[i];
  }
}

// logits = H @ co_w1 + b ; write sigmoid/softplus outputs for round r
__global__ void k_logits(const _Float16* __restrict__ H, const float* __restrict__ w,
                         const float* __restrict__ b, float* __restrict__ out, int r){
  for(size_t c=(size_t)blockIdx.x*blockDim.x+threadIdx.x; c<NC; c+=(size_t)gridDim.x*blockDim.x){
    float acc=b[0];
    #pragma unroll 8
    for(int k=0;k<128;k++) acc += (float)H[c*128+k]*w[k];
    out[(size_t)r*NC + c]                    = sigf(acc);
    out[(size_t)RNDS*NC + (size_t)r*NC + c]  = sppf(acc);
  }
}

// ---------------------------------------------------------------- WMMA GEMM
// C[M,N] = act(A[M,K](f16) @ W[K,N](f16) + bias), N in {128,256}, K%32==0, M%128==0.
// 256 threads = 8 wave32; wave w owns rows [blk*128+w*16, +16) x all N columns.
// Whole weight panel transposed into LDS (<=256KB, fits 320KB/WGP).
template<int N>
__global__ void k_gemm_wmma(const _Float16* __restrict__ A, const _Float16* __restrict__ W,
                            const float* __restrict__ bias, int M, int K,
                            float* __restrict__ outF, _Float16* __restrict__ outH, int relu)
{
  extern __shared__ _Float16 Wt[];                 // [N][K] transposed weights
  const int tid = threadIdx.x;
  for(int i=tid; i<K*N; i+=256){ int k=i/N, n=i%N; Wt[n*K+k]=W[i]; }
  __syncthreads();

  const int wave = tid>>5, lane = tid&31;
  const int lrow = lane&15, lhi = lane>>4;
  const int row0 = blockIdx.x*128 + wave*16;
  const _Float16* Arow = A + (size_t)(row0 + lrow)*K;

  v8f acc[N/16] = {};

  for(int k0=0; k0<K; k0+=32){
    __builtin_prefetch((const void*)(Arow + k0 + 128), 0, 1);   // global_prefetch_b8
    // A fragment 16x32 f16: lanes 0-15 -> K {0..7,16..23}; lanes 16-31 -> K {8..15,24..31}
    v16h a;
    {
      const v8h a0 = *(const v8h*)(Arow + k0      + lhi*8);
      const v8h a1 = *(const v8h*)(Arow + k0 + 16 + lhi*8);
      #pragma unroll
      for(int t=0;t<8;t++){ a[t]=a0[t]; a[8+t]=a1[t]; }
    }
    #pragma unroll
    for(int t=0;t<N/16;t++){
      // B fragment 32x16: lane col = lane&15, lanes 0-15 K {0..7,16..23}; lanes 16-31 K {8..15,24..31}
      const _Float16* Bp = Wt + (size_t)(t*16 + lrow)*K + k0 + lhi*8;
      const v8h b0 = *(const v8h*)(Bp);
      const v8h b1 = *(const v8h*)(Bp + 16);
      v16h b;
      #pragma unroll
      for(int u=0;u<8;u++){ b[u]=b0[u]; b[8+u]=b1[u]; }
      acc[t] = __builtin_amdgcn_wmma_f32_16x16x32_f16(false, a, false, b,
                                                      (short)0, acc[t], false, false);
    }
  }

  // C/D layout: VGPR j -> row = row0 + lhi*8 + j ; col = (lane&15) + t*16
  const int rb = row0 + lhi*8;
  #pragma unroll
  for(int t=0;t<N/16;t++){
    const int col = t*16 + lrow;
    const float bv = bias[col];
    #pragma unroll
    for(int j=0;j<8;j++){
      float v = acc[t][j] + bv;
      if(relu) v = fmaxf(v, 0.0f);
      size_t o = (size_t)(rb + j)*N + col;
      if(outH) outH[o]=(_Float16)v; else outF[o]=v;
    }
  }
}

// ---------------------------------------------------------------- host
static inline dim3 G(size_t n){ size_t b=(n+255)/256; if(b>((size_t)1<<20)) b=(size_t)1<<20; return dim3((unsigned)b); }

extern "C" void kernel_launch(void* const* d_in, const int* in_sizes, int n_in,
                              void* d_out, int out_size, void* d_ws, size_t ws_size,
                              hipStream_t stream)
{
  (void)in_sizes; (void)n_in; (void)out_size; (void)ws_size;
  const int*   edge_lit    = (const int*)d_in[0];
  const int*   edge_clause = (const int*)d_in[1];
  const float* noise_v=(const float*)d_in[2];  const float* noise_c=(const float*)d_in[3];
  const float* vq_w0=(const float*)d_in[4];    const float* vq_b0=(const float*)d_in[5];
  const float* vq_w1=(const float*)d_in[6];    const float* vq_b1=(const float*)d_in[7];
  const float* cq_w0=(const float*)d_in[8];    const float* cq_b0=(const float*)d_in[9];
  const float* cq_w1=(const float*)d_in[10];   const float* cq_b1=(const float*)d_in[11];
  const float* cm_w0=(const float*)d_in[12];   const float* cm_b0=(const float*)d_in[13];
  const float* cm_w1=(const float*)d_in[14];   const float* cm_b1=(const float*)d_in[15];
  const float* ug_w0=(const float*)d_in[16];   const float* ug_b0=(const float*)d_in[17];
  const float* ug_w1=(const float*)d_in[18];   const float* ug_b1=(const float*)d_in[19];
  const float* ug_w2=(const float*)d_in[20];   const float* ug_b2=(const float*)d_in[21];
  const float* co_w0=(const float*)d_in[22];   const float* co_b0=(const float*)d_in[23];
  const float* co_w1=(const float*)d_in[24];   const float* co_b1=(const float*)d_in[25];
  float* out=(float*)d_out;

  char* base=(char*)d_ws; size_t off=0;
  auto take=[&](size_t bytes)->char*{ char* p=base+off; off += (bytes+255)&~(size_t)255; return p; };

  _Float16* Wvq0=(_Float16*)take(160*128*2);  _Float16* Wvq1=(_Float16*)take(128*128*2);
  _Float16* Wcq0=(_Float16*)take(160*128*2);  _Float16* Wcq1=(_Float16*)take(128*128*2);
  _Float16* Wcm0=(_Float16*)take(384*256*2);  _Float16* Wcm1=(_Float16*)take(256*256*2);
  _Float16* Wug0=(_Float16*)take(512*256*2);  _Float16* Wug1=(_Float16*)take(256*256*2);
  _Float16* Wug2=(_Float16*)take(256*128*2);  _Float16* Wco0=(_Float16*)take(128*128*2);

  float* vars  =(float*)take((size_t)NV*128*4);
  float* cls   =(float*)take((size_t)NC*128*4);
  float* VQ    =(float*)take((size_t)NV*128*4);      // reused as ug-MLP output V
  float* csum  =(float*)take((size_t)NC*128*4);      // becomes dcsum in-place
  float* dlits =(float*)take((size_t)2*NV*128*4);
  float* vloss =(float*)take((size_t)2*NV*128*4);
  float* CD    =(float*)take((size_t)NC*256*4);      // CQ aliases first half
  float* CQ    = CD;
  float* deg   =(float*)take((size_t)2*NV*4);
  float* degw  =(float*)take((size_t)2*NV*4);
  float* vdw   =(float*)take((size_t)NV*4);
  float* stats =(float*)take(512*4);
  float* colsum=stats; float* sumsq=stats+128; float* ms=stats+132;

  _Float16* Xbuf=(_Float16*)take((size_t)NC*384*2);  // X_cq/X_vq(160) and X_cm(384) live here
  _Float16* Xco = Xbuf;                              // [NC,128]   (after X_cm consumed)
  _Float16* Xug = Xbuf + (size_t)NC*128;             // [NV,512]
  _Float16* H2  = Xbuf + (size_t)NC*256;             // [NV,256]
  _Float16* Hbuf=(_Float16*)take((size_t)NC*256*2);  // hidden activations

  auto gemm128=[&](const _Float16* A,const _Float16* W,const float* b,int M,int K,
                   float* oF,_Float16* oH,int relu){
    k_gemm_wmma<128><<<dim3(M/128),dim3(256),(size_t)K*128*2,stream>>>(A,W,b,M,K,oF,oH,relu);
  };
  auto gemm256=[&](const _Float16* A,const _Float16* W,const float* b,int M,int K,
                   float* oF,_Float16* oH,int relu){
    k_gemm_wmma<256><<<dim3(M/128),dim3(256),(size_t)K*256*2,stream>>>(A,W,b,M,K,oF,oH,relu);
  };

  // ---- one-time (per call) prep
  k_w2h<<<G(160*128),256,0,stream>>>(vq_w0,Wvq0,132,160,128);
  k_w2h<<<G(128*128),256,0,stream>>>(vq_w1,Wvq1,128,128,128);
  k_w2h<<<G(160*128),256,0,stream>>>(cq_w0,Wcq0,132,160,128);
  k_w2h<<<G(128*128),256,0,stream>>>(cq_w1,Wcq1,128,128,128);
  k_w2h<<<G(384*256),256,0,stream>>>(cm_w0,Wcm0,384,384,256);
  k_w2h<<<G(256*256),256,0,stream>>>(cm_w1,Wcm1,256,256,256);
  k_w2h<<<G(512*256),256,0,stream>>>(ug_w0,Wug0,512,512,256);
  k_w2h<<<G(256*256),256,0,stream>>>(ug_w1,Wug1,256,256,256);
  k_w2h<<<G(256*128),256,0,stream>>>(ug_w2,Wug2,256,256,128);
  k_w2h<<<G(128*128),256,0,stream>>>(co_w0,Wco0,128,128,128);

  k_fill<<<G((size_t)2*NV),256,0,stream>>>(deg,0.0f,(size_t)2*NV);
  k_deg <<<G(NE),256,0,stream>>>(edge_lit,deg);
  k_degw<<<G((size_t)2*NV),256,0,stream>>>(deg,degw,vdw);
  k_fill<<<G((size_t)NV*128),256,0,stream>>>(vars,1.0f,(size_t)NV*128);
  k_fill<<<G((size_t)NC*128),256,0,stream>>>(cls ,1.0f,(size_t)NC*128);

  for(int r=0;r<RNDS;r++){
    const float* nv = noise_v + (size_t)r*NV*4;
    const float* nc = noise_c + (size_t)r*NC*4;

    // vq = mlp2([variables|nv])
    k_build_xin<<<G((size_t)NV*160),256,0,stream>>>(vars,nv,Xbuf,NV);
    gemm128(Xbuf,Wvq0,vq_b0,NV,160,nullptr,Hbuf,1);
    gemm128(Hbuf,Wvq1,vq_b1,NV,128,VQ,nullptr,0);

    // cq = mlp2([clauses|nc])
    k_build_xin<<<G((size_t)NC*160),256,0,stream>>>(cls,nc,Xbuf,NC);
    gemm128(Xbuf,Wcq0,cq_b0,NC,160,nullptr,Hbuf,1);
    gemm128(Hbuf,Wcq1,cq_b1,NC,128,CQ,nullptr,0);

    // csum = segment_sum(softplus(+-vq))
    k_fill<<<G((size_t)NC*128),256,0,stream>>>(csum,0.0f,(size_t)NC*128);
    k_scatter_lits<<<G((size_t)NE*128),256,0,stream>>>(edge_lit,edge_clause,VQ,csum);

    // loss, cgrad, dcsum ; build X_cm = [clauses | 4*loss | cgrad]
    k_clause_mid<<<G((size_t)NC*128),256,0,stream>>>(csum,CQ,cls,Xbuf);

    // dlits = segment_sum(dcsum over edges)
    k_fill<<<G((size_t)2*NV*128),256,0,stream>>>(dlits,0.0f,(size_t)2*NV*128);
    k_scatter_gather<<<G((size_t)NE*128),256,0,stream>>>(edge_clause,edge_lit,csum,128,dlits);

    // clause message MLP
    gemm256(Xbuf,Wcm0,cm_b0,NC,384,nullptr,Hbuf,1);
    gemm256(Hbuf,Wcm1,cm_b1,NC,256,CD,nullptr,0);

    // pair_norm over CD[:,128:256]
    k_fill<<<1,256,0,stream>>>(stats,0.0f,129);
    k_pn_reduce<<<dim3(2048),dim3(128),0,stream>>>(CD,NC,256,128,colsum,sumsq);
    k_pn_final <<<1,128,0,stream>>>(colsum,sumsq,1.0f/(float)NC,ms);

    // variables_loss = segment_sum(CD[:,0:128])
    k_fill<<<G((size_t)2*NV*128),256,0,stream>>>(vloss,0.0f,(size_t)2*NV*128);
    k_scatter_gather<<<G((size_t)NE*128),256,0,stream>>>(edge_clause,edge_lit,CD,256,vloss);

    // clauses update (+ f16 copy for the clause-output MLP)
    k_apply_clause<<<G((size_t)NC*128),256,0,stream>>>(CD,ms,cls,Xco);

    // unit = [vgrad | variables | vloss_pos*dw | vloss_neg*dw] ; ug MLP3
    k_build_ug<<<G((size_t)NV*128),256,0,stream>>>(dlits,VQ,vdw,vars,vloss,degw,Xug);
    gemm256(Xug ,Wug0,ug_b0,NV,512,nullptr,Hbuf,1);
    gemm256(Hbuf,Wug1,ug_b1,NV,256,nullptr,H2 ,1);
    gemm128(H2  ,Wug2,ug_b2,NV,256,VQ,nullptr,0);          // V reuses VQ storage

    // pair_norm + variables update
    k_fill<<<1,256,0,stream>>>(stats,0.0f,129);
    k_pn_reduce<<<dim3(2048),dim3(128),0,stream>>>(VQ,NV,128,0,colsum,sumsq);
    k_pn_final <<<1,128,0,stream>>>(colsum,sumsq,1.0f/(float)NV,ms);
    k_apply_var<<<G((size_t)NV*128),256,0,stream>>>(VQ,ms,vars);

    // clause output head
    gemm128(Xco,Wco0,co_b0,NC,128,nullptr,Hbuf,1);
    k_logits<<<G(NC),256,0,stream>>>(Hbuf,co_w1,co_b1,out,r);
  }
}
